// RNN_31267361915159
// MI455X (gfx1250) — matrix operand order repaired and checked by reading, not compiled, not run on hardware
//
#include <hip/hip_runtime.h>
#include <math.h>

#define SEQ_LEN 2048
#define BATCH   256
#define DIM     128     // in_dim == hidden_dim == out_dim == 128
#define HSTR    136     // padded LDS row stride (bf16): 272 B = 68 dwords -> lane m
                        // starts at bank 4m mod 64: 16 lanes tile all 64 banks.
#define PFD     5       // x prefetch depth in LDS slots

typedef __attribute__((ext_vector_type(16))) __bf16       bf16x16;
typedef __attribute__((ext_vector_type(8)))  __bf16       bf16x8;
typedef __attribute__((ext_vector_type(8)))  float        f32x8;
typedef __attribute__((ext_vector_type(4)))  float        f32x4;
typedef __attribute__((ext_vector_type(4)))  unsigned int u32x4;

__device__ __forceinline__ float tanh_f32(float x) {
#if __has_builtin(__builtin_amdgcn_tanhf)
  return __builtin_amdgcn_tanhf(x);   // v_tanh_f32 (TRANS op on gfx1250)
#else
  return tanhf(x);
#endif
}

__device__ __forceinline__ void sched_fence() {
#if __has_builtin(__builtin_amdgcn_sched_barrier)
  __builtin_amdgcn_sched_barrier(0);  // pin: all prior loads issue before WMMAs
#endif
}

// Wait until this wave's outstanding async (global<->LDS) ops <= n.
#define WAIT_ASYNCCNT(n) asm volatile("s_wait_asynccnt %0" ::"i"(n) : "memory")

// Async-stage one step's x tile (16 rows x 128 f32 = 8 KB, contiguous in x) into
// an LDS slot. 256 threads x 2 x b128 = 8 KB. Tracked by ASYNCcnt (2 per wave).
__device__ __forceinline__ void async_stage_x(const float* gsrc, unsigned ldsOff,
                                              int tid) {
  unsigned o0 = (unsigned)tid * 16u;               // 0..4080
  unsigned long long g0 =
      (unsigned long long)(uintptr_t)gsrc + (unsigned long long)o0;
  unsigned long long g1 = g0 + 4096ull;
  unsigned l0 = ldsOff + o0;
  unsigned l1 = l0 + 4096u;
  asm volatile("global_load_async_to_lds_b128 %0, %1, off"
               ::"v"(l0), "v"(g0) : "memory");
  asm volatile("global_load_async_to_lds_b128 %0, %1, off"
               ::"v"(l1), "v"(g1) : "memory");
}

// Cooperative f32 -> bf16 conversion of one staged x tile (2048 elements) into
// the bank-padded bf16 layout. Each thread: 8 contiguous elements of one row.
__device__ __forceinline__ void convert_x_slot(const float* src, __bf16* dst,
                                               int tid) {
  const int e0 = tid * 8;
  const int m  = e0 >> 7;        // row 0..15
  const int k  = e0 & 127;       // column within row
  f32x4 a = *(const f32x4*)(src + e0);
  f32x4 b = *(const f32x4*)(src + e0 + 4);
  bf16x8 o;
#pragma unroll
  for (int e = 0; e < 4; ++e) {
    o[e]     = (__bf16)a[e];
    o[4 + e] = (__bf16)b[e];
  }
  *(bf16x8*)(dst + m * HSTR + k) = o;
}

// 16x32 bf16 A-fragment from a padded row-major bf16 matrix in LDS (2 b128).
// A layout (ISA 7.12.2): lanes 0-15 hold K {0..7,16..23}, lanes 16-31 {8..15,24..31}.
__device__ __forceinline__ bf16x16 load_a_lds_bf16(const __bf16* h, int m, int klo) {
  const __bf16* p = h + m * HSTR + klo;
  u32x4 lo = *(const u32x4*)(p);
  u32x4 hi = *(const u32x4*)(p + 16);
  bf16x16 r;
  ((u32x4*)&r)[0] = lo;
  ((u32x4*)&r)[1] = hi;
  return r;
}

// 32x16 bf16 B-fragment (column block `col`) from row-major f32 W[DIM][DIM].
// B layout: lane<16 holds K {0..7,16..23} of N=lane (2/VGPR), lane>=16 holds K {8..15,24..31}.
__device__ __forceinline__ bf16x16 load_b_global_f32(const float* W, int kc, int kb,
                                                     int col) {
  const float* p = W + (kc * 32 + (kb >> 1)) * DIM + col;
  bf16x16 r;
#pragma unroll
  for (int e = 0; e < 16; ++e) r[e] = (__bf16)p[(e + (e >= 8 ? 8 : 0)) * DIM];
  return r;
}

#define WMMA_BF16(a, b, c) \
  __builtin_amdgcn_wmma_f32_16x16x32_bf16(false, (a), false, (b), (short)0, (c), \
                                          false, false)

__global__ __launch_bounds__(256, 1)
void rnn_scan_wmma(const float* __restrict__ x,
                   const float* __restrict__ W_xh,
                   const float* __restrict__ W_hh,
                   const float* __restrict__ b_h,
                   const float* __restrict__ W_hy,
                   const float* __restrict__ b_y,
                   float* __restrict__ out) {
  // x staging ring: PFD slots of [16 rows x 128 f32] = 8 KB each (DMA layout).
  __shared__ float  xstage[PFD][16 * DIM];
  // Ping-pong bf16 x tile (bank-padded) for compute / prepare.
  __shared__ __bf16 xbf[2][16 * HSTR];
  // Double-buffered hidden state, bf16 bank-padded row-major [16 x HSTR].
  __shared__ __bf16 hbuf[2][16 * HSTR];

  const int tid  = threadIdx.x;
  const int lane = tid & 31;
  const int wave = tid >> 5;            // N-tile 0..7 (16 columns each)
  const int hihf = lane >> 4;           // 0: lanes 0-15, 1: lanes 16-31
  const int lm   = lane & 15;

  const int klo  = hihf ? 8 : 0;        // A-fragment K sub-base
  const int kb   = hihf ? 16 : 0;       // B-fragment K sub-base
  const int ncol = wave * 16 + lm;      // output column this lane owns (C/D layout)
  const int mrow = lm;                  // A row this lane supplies
  const int mOff = hihf ? 8 : 0;        // C/D M offset for upper half-wave

  const int brow0 = blockIdx.x * 16;    // batch tile base row

  const unsigned xbase = (unsigned)(uintptr_t)&xstage[0][0];  // LDS byte offset

  // ---- one-time: bf16 weight fragments pinned in VGPRs for the whole scan ----
  bf16x16 Bxh[4], Bhh[4];
#pragma unroll
  for (int kc = 0; kc < 4; ++kc) {
    Bxh[kc] = load_b_global_f32(W_xh, kc, kb, ncol);
    Bhh[kc] = load_b_global_f32(W_hh, kc, kb, ncol);
  }
  // Bias pre-broadcast into a WMMA C operand: bias add costs zero VALU in-loop.
  f32x8 bhv;
  {
    const float bh = b_h[ncol];
#pragma unroll
    for (int v = 0; v < 8; ++v) bhv[v] = bh;
  }

  // h0 = 0 (zero the whole padded buffer)
  for (int i = tid; i < 16 * HSTR; i += 256) hbuf[0][i] = (__bf16)0.0f;

  // Prologue: stage x for steps 0..PFD-1.
#pragma unroll
  for (int p = 0; p < PFD; ++p)
    async_stage_x(x + ((size_t)p * BATCH + brow0) * DIM,
                  xbase + (unsigned)p * (16 * DIM * 4), tid);
  WAIT_ASYNCCNT(2 * (PFD - 1));       // set 0 complete (this wave)
  __syncthreads();                    // h0 + all waves' set-0 bytes visible
  convert_x_slot(xstage[0], xbf[0], tid);
  WAIT_ASYNCCNT(2 * (PFD - 2));       // set 1 complete (this wave)
  __syncthreads();                    // xbf[0] + set-1 bytes visible

  __bf16* cur = hbuf[0];
  __bf16* nxt = hbuf[1];

#pragma unroll 1
  for (int t = 0; t < SEQ_LEN; ++t) {
    const __bf16* xb = xbf[t & 1];

    // Issue ALL 16 fragment loads before any WMMA consumes (sched fence below).
    bf16x16 h0 = load_a_lds_bf16(cur, mrow, 0 * 32 + klo);
    bf16x16 h1 = load_a_lds_bf16(cur, mrow, 1 * 32 + klo);
    bf16x16 h2 = load_a_lds_bf16(cur, mrow, 2 * 32 + klo);
    bf16x16 h3 = load_a_lds_bf16(cur, mrow, 3 * 32 + klo);
    bf16x16 x0 = load_a_lds_bf16(xb, mrow, 0 * 32 + klo);
    bf16x16 x1 = load_a_lds_bf16(xb, mrow, 1 * 32 + klo);
    bf16x16 x2 = load_a_lds_bf16(xb, mrow, 2 * 32 + klo);
    bf16x16 x3 = load_a_lds_bf16(xb, mrow, 3 * 32 + klo);
    sched_fence();

    // Two independent accumulator chains; acc0 starts from the bias vector.
    f32x8 acc0 = WMMA_BF16(x0, Bxh[0], bhv);
    f32x8 acc1 = {};
    acc1 = WMMA_BF16(x1, Bxh[1], acc1);
    acc0 = WMMA_BF16(x2, Bxh[2], acc0);
    acc1 = WMMA_BF16(x3, Bxh[3], acc1);
    acc0 = WMMA_BF16(h0, Bhh[0], acc0);
    acc1 = WMMA_BF16(h1, Bhh[1], acc1);
    acc0 = WMMA_BF16(h2, Bhh[2], acc0);
    acc1 = WMMA_BF16(h3, Bhh[3], acc1);

    // h_new = tanh(acc0 + acc1); write bf16 into the other buffer.
#pragma unroll
    for (int v = 0; v < 8; ++v) {
      float hv = tanh_f32(acc0[v] + acc1[v]);
      nxt[(v + mOff) * HSTR + ncol] = (__bf16)hv;
    }

    // Prepare next step's bf16 x tile (slot t+1 is complete; independent of h).
    if (t + 1 < SEQ_LEN)
      convert_x_slot(xstage[(t + 1) % PFD], xbf[(t + 1) & 1], tid);

    // Complete this wave's async set for step t+2 BEFORE the barrier, so step
    // t+1's conversion (and step t+1's h reads) are safe after the barrier.
    WAIT_ASYNCCNT(2 * (PFD - 3));
    __syncthreads();

    // Slot t%PFD was last read by the conversion in step t-1 -> safe to refill.
    if (t + PFD < SEQ_LEN)
      async_stage_x(x + ((size_t)(t + PFD) * BATCH + brow0) * DIM,
                    xbase + (unsigned)(t % PFD) * (16 * DIM * 4), tid);

    __bf16* tmp = cur; cur = nxt; nxt = tmp;
  }

  // ---- out = h_final @ W_hy + b_y ----
  bf16x16 Bhy[4];
#pragma unroll
  for (int kc = 0; kc < 4; ++kc) Bhy[kc] = load_b_global_f32(W_hy, kc, kb, ncol);

  f32x8 o;
  {
    const float by = b_y[ncol];
#pragma unroll
    for (int v = 0; v < 8; ++v) o[v] = by;
  }
#pragma unroll
  for (int kc = 0; kc < 4; ++kc) {
    bf16x16 ha = load_a_lds_bf16(cur, mrow, kc * 32 + klo);
    o = WMMA_BF16(ha, Bhy[kc], o);
  }
#pragma unroll
  for (int v = 0; v < 8; ++v)
    out[(size_t)(brow0 + v + mOff) * DIM + ncol] = o[v];
}

extern "C" void kernel_launch(void* const* d_in, const int* in_sizes, int n_in,
                              void* d_out, int out_size, void* d_ws, size_t ws_size,
                              hipStream_t stream) {
  const float* x   = (const float*)d_in[0];
  const float* Wxh = (const float*)d_in[1];
  const float* Whh = (const float*)d_in[2];
  const float* bh  = (const float*)d_in[3];
  const float* Why = (const float*)d_in[4];
  const float* by  = (const float*)d_in[5];
  float* out = (float*)d_out;

  rnn_scan_wmma<<<BATCH / 16, 256, 0, stream>>>(x, Wxh, Whh, bh, Why, by, out);
}
